// cliqueMPNN_hindsight_7481833029835
// MI455X (gfx1250) — matrix operand order repaired and checked by reading, not compiled, not run on hardware
//
#include <hip/hip_runtime.h>
#include <hip/hip_bf16.h>
#include <math.h>

typedef __attribute__((ext_vector_type(16))) _Float16 v16h;
typedef __attribute__((ext_vector_type(8)))  float    v8f;
typedef __attribute__((ext_vector_type(4)))  float    v4f;

#define THREADS 256
#define CEILDIV(a,b) (((a)+(b)-1)/(b))

__device__ __forceinline__ float leakyf(float x, float s) { return x > 0.f ? x : s * x; }

// Monotone order-preserving float->uint mapping for atomic max/min on floats.
__device__ __forceinline__ unsigned flipf(float f) {
  unsigned u = __float_as_uint(f);
  return (u & 0x80000000u) ? ~u : (u | 0x80000000u);
}
__device__ __forceinline__ float unflipf(unsigned u) {
  return (u & 0x80000000u) ? __uint_as_float(u ^ 0x80000000u) : __uint_as_float(~u);
}

// ---------------------------------------------------------------------------
// Pack W[K,Nout] (f32, row-major) into per-lane WMMA B-fragment layout (f16):
//   Wf[(kc*tilesN + tileN)*32 + lane] = v16h fragment for that lane.
// B 32x16 layout: N = lane%16; lanes 0-15 hold K=kc*32+{0..7,16..23}, lanes 16-31 +{8..15,24..31}.
// ---------------------------------------------------------------------------
__global__ void k_pack_w(const float* __restrict__ W, _Float16* __restrict__ Wf,
                         int K, int Nout) {
  const int tilesN = Nout >> 4;
  const int total  = (K >> 5) * tilesN * 32;
  int id = blockIdx.x * blockDim.x + threadIdx.x;
  if (id >= total) return;
  const int lane  = id & 31;
  const int t     = id >> 5;
  const int tileN = t % tilesN;
  const int kc    = t / tilesN;
  const int half  = lane >> 4;
  const int l16   = lane & 15;
  const int bn    = (tileN << 4) + l16;
  const int bk0   = (kc << 5) + (half << 3);
  _Float16* dst = Wf + (size_t)id * 16;
#pragma unroll
  for (int j = 0; j < 16; ++j)
    dst[j] = (_Float16)W[(size_t)(bk0 + j + ((j >> 3) << 3)) * Nout + bn];
}

// ---------------------------------------------------------------------------
// WMMA GEMM: C[M,Nout] = act(A[M,K] @ W[K,Nout] + bias)
// f32 storage, f16 WMMA compute, f32 accumulate (v_wmma_f32_16x16x32_f16).
// Requires M % 16 == 0, K % 32 == 0, Nout % 64 == 0.
// One 16x64 output strip per wave32 (4 accumulators, 4 WMMAs per K-chunk).
// W pre-packed into B-fragment layout by k_pack_w.
// ACT: 0 = none, 1 = relu, 2 = leaky(0.01)
// ---------------------------------------------------------------------------
template<int ACT>
__global__ void wmma_gemm_kernel(const float* __restrict__ A,
                                 const _Float16* __restrict__ Wf,
                                 const float* __restrict__ bias,
                                 float* __restrict__ C,
                                 int M, int K, int Nout) {
  const int gwave   = (blockIdx.x * blockDim.x + threadIdx.x) >> 5;
  const int lane    = threadIdx.x & 31;
  const int groupsN = Nout >> 6;              // 4 N-tiles per wave
  const int tilesN  = Nout >> 4;
  const int tilesM  = M >> 4;                 // M always a multiple of 16 here
  const int tileM   = gwave / groupsN;
  const int grpN    = gwave - tileM * groupsN;
  if (tileM >= tilesM) return;                // wave-uniform: EXEC stays all-ones
  const int m0   = tileM << 4;
  const int half = lane >> 4;                 // 0: lanes 0-15, 1: lanes 16-31
  const int l16  = lane & 15;

  const float* __restrict__ arow = A + (size_t)(m0 + l16) * K;
  const v16h*  __restrict__ Wfv  = (const v16h*)Wf;
  const int nChunks = K >> 5;

  v8f acc[4] = {v8f{}, v8f{}, v8f{}, v8f{}};
  for (int kc = 0; kc < nChunks; ++kc) {
    // A fragment: 16-bit 16x32 layout (ISA 7.12.2); two contiguous 16B runs.
    const float* ap = arow + (kc << 5) + (half << 3);
    const v4f a0 = *(const v4f*)(ap);
    const v4f a1 = *(const v4f*)(ap + 4);
    const v4f a2 = *(const v4f*)(ap + 16);
    const v4f a3 = *(const v4f*)(ap + 20);
    v16h a;
#pragma unroll
    for (int j = 0; j < 4; ++j) {
      a[j]      = (_Float16)a0[j];
      a[j + 4]  = (_Float16)a1[j];
      a[j + 8]  = (_Float16)a2[j];
      a[j + 12] = (_Float16)a3[j];
    }
    // B fragments: one 32-byte vector load each, pre-packed layout.
    const v16h* bp = Wfv + ((size_t)kc * tilesN + (grpN << 2)) * 32 + lane;
    const v16h b0 = bp[0];
    const v16h b1 = bp[32];
    const v16h b2 = bp[64];
    const v16h b3 = bp[96];
    acc[0] = __builtin_amdgcn_wmma_f32_16x16x32_f16(false, a, false, b0, (short)0, acc[0], false, false);
    acc[1] = __builtin_amdgcn_wmma_f32_16x16x32_f16(false, a, false, b1, (short)0, acc[1], false, false);
    acc[2] = __builtin_amdgcn_wmma_f32_16x16x32_f16(false, a, false, b2, (short)0, acc[2], false, false);
    acc[3] = __builtin_amdgcn_wmma_f32_16x16x32_f16(false, a, false, b3, (short)0, acc[3], false, false);
  }

  const int mbase = m0 + (half << 3);          // C/D: VGPR r -> row r + 8*(lane/16)
#pragma unroll
  for (int t = 0; t < 4; ++t) {
    const int n  = ((grpN << 2) + t) * 16 + l16;
    const float bb = (bias != nullptr) ? bias[n] : 0.f;
#pragma unroll
    for (int r = 0; r < 8; ++r) {
      float v = acc[t][r] + bb;
      if (ACT == 1) v = fmaxf(v, 0.f);
      if (ACT == 2) v = leakyf(v, 0.01f);
      C[(size_t)(mbase + r) * Nout + n] = v;
    }
  }
}

// ---------------------------------------------------------------------------
// Elementwise / scatter kernels
// ---------------------------------------------------------------------------
__global__ void k_prop_scalar(const int* __restrict__ row, const int* __restrict__ col,
                              const float* __restrict__ src, float* __restrict__ dst, int E) {
  int e = blockIdx.x * blockDim.x + threadIdx.x;
  if (e >= E) return;
  atomicAdd(&dst[row[e]], src[col[e]]);
}

__global__ void k_thresh(float* __restrict__ mask, const float* __restrict__ s, int n) {
  int i = blockIdx.x * blockDim.x + threadIdx.x;
  if (i >= n) return;
  mask[i] = (s[i] > 0.f) ? 1.f : 0.f;
}

// h1[n,j] = relu((propx[n] + x[n]) * w1[j] + b1[j])
__global__ void k_conv1_l1(float* __restrict__ out, const float* __restrict__ scal,
                           const float* __restrict__ x, const float* __restrict__ w1,
                           const float* __restrict__ b1, int total) {
  int id = blockIdx.x * blockDim.x + threadIdx.x;
  if (id >= total) return;
  int n = id >> 6, j = id & 63;
  float h0 = scal[n] + x[n];
  out[id] = fmaxf(h0 * w1[j] + b1[j], 0.f);
}

// agg = (1 + eps[layer]) * h
__global__ void k_scale_init(float* __restrict__ agg, const float* __restrict__ h,
                             const float* __restrict__ eps, int layer, int total) {
  int id = blockIdx.x * blockDim.x + threadIdx.x;
  if (id >= total) return;
  agg[id] = (1.f + eps[layer]) * h[id];
}

// agg[row[e],f] += h[col[e],f]   (F = 64, feature-major for coalescing)
__global__ void k_prop_feat(const int* __restrict__ row, const int* __restrict__ col,
                            const float* __restrict__ h, float* __restrict__ agg, int E) {
  int id = blockIdx.x * blockDim.x + threadIdx.x;
  if (id >= E * 64) return;
  int e = id >> 6, f = id & 63;
  atomicAdd(&agg[(size_t)row[e] * 64 + f], h[(size_t)col[e] * 64 + f]);
}

__global__ void k_add(float* __restrict__ h, const float* __restrict__ t, int total) {
  int id = blockIdx.x * blockDim.x + threadIdx.x;
  if (id >= total) return;
  h[id] += t[id];
}

// mode 0: bn only; mode 1: bn then *mask; mode 2: *mask then bn
__global__ void k_bn(float* __restrict__ h, const float* __restrict__ p,
                     const float* __restrict__ mask, int mode, int N, int F) {
  int id = blockIdx.x * blockDim.x + threadIdx.x;
  if (id >= N * F) return;
  int n = id / F, j = id % F;
  float g = p[j], b = p[F + j], m = p[2 * F + j], v = p[3 * F + j];
  float hv = h[id];
  if (mode == 2 && mask) hv *= mask[n];
  hv = (hv - m) * rsqrtf(v + 1e-5f) * g + b;
  if (mode == 1 && mask) hv *= mask[n];
  h[id] = hv;
}

// Per-node per-head attention dot products (factorized GAT attention)
__global__ void k_gat_dots(const float* __restrict__ xt, const float* __restrict__ att,
                           float* __restrict__ asrc, float* __restrict__ adst, int N) {
  int id = blockIdx.x * blockDim.x + threadIdx.x;
  if (id >= N * 8) return;
  int n = id >> 3, hd = id & 7;
  const float* xp = xt + (size_t)n * 256 + hd * 32;
  const float* ap = att + hd * 64;
  float s1 = 0.f, s2 = 0.f;
#pragma unroll
  for (int k = 0; k < 32; ++k) { s1 += xp[k] * ap[k]; s2 += xp[k] * ap[32 + k]; }
  asrc[id] = s1;
  adst[id] = s2;
}

__global__ void k_gat_alpha1(const int* __restrict__ row, const int* __restrict__ col,
                             const float* __restrict__ asrc, const float* __restrict__ adst,
                             float* __restrict__ alpha, unsigned* __restrict__ amax,
                             int E, int Etot) {
  int id = blockIdx.x * blockDim.x + threadIdx.x;
  if (id >= Etot * 8) return;
  int e = id >> 3, hd = id & 7;
  int r = (e < E) ? row[e] : (e - E);
  int c = (e < E) ? col[e] : (e - E);
  float a = asrc[r * 8 + hd] + adst[c * 8 + hd];
  a = leakyf(a, 0.2f);
  alpha[id] = a;
  atomicMax(&amax[r * 8 + hd], flipf(a));
}

__global__ void k_gat_alpha2(const int* __restrict__ row, float* __restrict__ alpha,
                             const unsigned* __restrict__ amax, float* __restrict__ denom,
                             int E, int Etot) {
  int id = blockIdx.x * blockDim.x + threadIdx.x;
  if (id >= Etot * 8) return;
  int e = id >> 3, hd = id & 7;
  int r = (e < E) ? row[e] : (e - E);
  float a = expf(alpha[id] - unflipf(amax[r * 8 + hd]));
  alpha[id] = a;
  atomicAdd(&denom[r * 8 + hd], a);
}

// alpha /= (denom[frow] + 1e-16)  (done once per (edge, head))
__global__ void k_gat_norm(const int* __restrict__ row, float* __restrict__ alpha,
                           const float* __restrict__ denom, int E, int Etot) {
  int id = blockIdx.x * blockDim.x + threadIdx.x;
  if (id >= Etot * 8) return;
  int e = id >> 3, hd = id & 7;
  int r = (e < E) ? row[e] : (e - E);
  alpha[id] = alpha[id] / (denom[r * 8 + hd] + 1e-16f);
}

__global__ void k_gat_agg(const int* __restrict__ row, const int* __restrict__ col,
                          const float* __restrict__ alpha,
                          const float* __restrict__ xt, float* __restrict__ gout,
                          int E, int Etot) {
  int id = blockIdx.x * blockDim.x + threadIdx.x;
  if (id >= Etot * 256) return;
  int e = id >> 8, c = id & 255, hd = c >> 5;
  int r  = (e < E) ? row[e] : (e - E);
  int cn = (e < E) ? col[e] : (e - E);
  float a = alpha[e * 8 + hd];
  atomicAdd(&gout[(size_t)r * 256 + c], a * xt[(size_t)cn * 256 + c]);
}

__global__ void k_bias_mask(float* __restrict__ h, const float* __restrict__ bias,
                            const float* __restrict__ mask, int N) {
  int id = blockIdx.x * blockDim.x + threadIdx.x;
  if (id >= N * 256) return;
  int n = id >> 8, j = id & 255;
  h[id] = (h[id] + bias[j]) * mask[n];
}

__global__ void k_lin2(const float* __restrict__ t, const float* __restrict__ w,
                       const float* __restrict__ b, const float* __restrict__ mask,
                       float* __restrict__ hl, int N) {
  int n = blockIdx.x * blockDim.x + threadIdx.x;
  if (n >= N) return;
  const float* tp = t + (size_t)n * 64;
  float s = 0.f;
#pragma unroll
  for (int j = 0; j < 64; ++j) s += tp[j] * w[j];
  hl[n] = leakyf(s + b[0], 0.01f) * mask[n];
}

__global__ void k_group_minmax(const float* __restrict__ hl, const int* __restrict__ batch,
                               unsigned* __restrict__ gmax, unsigned* __restrict__ gmin, int N) {
  int n = blockIdx.x * blockDim.x + threadIdx.x;
  if (n >= N) return;
  unsigned f = flipf(hl[n]);
  int g = batch[n];
  atomicMax(&gmax[g], f);
  atomicMin(&gmin[g], f);
}

__global__ void k_final(const float* __restrict__ hl, const int* __restrict__ batch,
                        const unsigned* __restrict__ gmax, const unsigned* __restrict__ gmin,
                        const float* __restrict__ mask, float* __restrict__ out, int N) {
  int n = blockIdx.x * blockDim.x + threadIdx.x;
  if (n >= N) return;
  int g = batch[n];
  float bmax = unflipf(gmax[g]);
  float bmin = unflipf(gmin[g]);
  float v = (hl[n] - bmin) / (bmax + 1e-6f - bmin);
  out[n] = v * mask[n] + mask[n] * 1e-6f;
}

// ---------------------------------------------------------------------------
extern "C" void kernel_launch(void* const* d_in, const int* in_sizes, int n_in,
                              void* d_out, int out_size, void* d_ws, size_t ws_size,
                              hipStream_t stream) {
  (void)n_in; (void)out_size; (void)ws_size;
  const float* x     = (const float*)d_in[0];
  const int*   row   = (const int*)d_in[1];
  const int    N     = in_sizes[0];
  const int    E     = in_sizes[1] / 2;
  const int*   col   = row + E;
  const int*   batch = (const int*)d_in[2];
  const float* c1w1  = (const float*)d_in[3];
  const float* c1b1  = (const float*)d_in[4];
  const float* c1w2  = (const float*)d_in[5];
  const float* c1b2  = (const float*)d_in[6];
  const float* c1bn  = (const float*)d_in[7];
  const float* gw1   = (const float*)d_in[8];
  const float* gb1   = (const float*)d_in[9];
  const float* gw2   = (const float*)d_in[10];
  const float* gb2   = (const float*)d_in[11];
  const float* gbn   = (const float*)d_in[12];
  const float* geps  = (const float*)d_in[13];
  const float* bnsbn = (const float*)d_in[14];
  const float* gatw  = (const float*)d_in[15];
  const float* gatt  = (const float*)d_in[16];
  const float* gatb  = (const float*)d_in[17];
  const float* l1w   = (const float*)d_in[18];
  const float* l1b   = (const float*)d_in[19];
  const float* bn2   = (const float*)d_in[20];
  const float* l2w   = (const float*)d_in[21];
  const float* l2b   = (const float*)d_in[22];
  const int L    = in_sizes[13];
  const int Etot = E + N;

  // ---- workspace layout (Wf first: 32B alignment for v16h loads) ----
  _Float16* Wf = (_Float16*)d_ws;                 // 16384 halves (32 KB) max
  float* ws   = (float*)d_ws + 8192;              // 16384 halves == 8192 floats
  float* h    = ws;  ws += (size_t)N * 64;
  float* agg  = ws;  ws += (size_t)N * 64;
  float* tbuf = ws;  ws += (size_t)N * 64;
  float* mask = ws;  ws += N;
  float* scal = ws;  ws += N;
  float* xt   = ws;  ws += (size_t)N * 256;
  float* gout = ws;  ws += (size_t)N * 256;
  float* asrc = ws;  ws += (size_t)N * 8;
  float* adst = ws;  ws += (size_t)N * 8;
  float* alpha= ws;  ws += (size_t)Etot * 8;
  unsigned* amax = (unsigned*)ws;  ws += (size_t)N * 8;
  float* denom= ws;  ws += (size_t)N * 8;
  float* hl   = ws;  ws += N;
  unsigned* gmax = (unsigned*)ws;  ws += 32;
  unsigned* gmin = (unsigned*)ws;  ws += 32;

  auto gemm = [&](const float* A, const float* W, const float* b, float* Cm,
                  int M, int K, int Nout, int act) {
    int packTotal = (K / 32) * (Nout / 16) * 32;
    k_pack_w<<<CEILDIV(packTotal, THREADS), THREADS, 0, stream>>>(W, Wf, K, Nout);
    int tiles  = (M / 16) * (Nout / 64);
    int blocks = CEILDIV(tiles, THREADS / 32);
    if (act == 1)
      wmma_gemm_kernel<1><<<blocks, THREADS, 0, stream>>>(A, Wf, b, Cm, M, K, Nout);
    else if (act == 2)
      wmma_gemm_kernel<2><<<blocks, THREADS, 0, stream>>>(A, Wf, b, Cm, M, K, Nout);
    else
      wmma_gemm_kernel<0><<<blocks, THREADS, 0, stream>>>(A, Wf, b, Cm, M, K, Nout);
  };
  auto prop_mask = [&]() {
    hipMemsetAsync(scal, 0, (size_t)N * sizeof(float), stream);
    k_prop_scalar<<<CEILDIV(E, THREADS), THREADS, 0, stream>>>(row, col, mask, scal, E);
    k_thresh<<<CEILDIV(N, THREADS), THREADS, 0, stream>>>(mask, scal, N);
  };

  // ---- conv1 + initial mask ----
  hipMemsetAsync(scal, 0, (size_t)N * sizeof(float), stream);
  k_prop_scalar<<<CEILDIV(E, THREADS), THREADS, 0, stream>>>(row, col, x, scal, E);
  k_thresh<<<CEILDIV(N, THREADS), THREADS, 0, stream>>>(mask, scal, N);
  k_conv1_l1<<<CEILDIV(N * 64, THREADS), THREADS, 0, stream>>>(agg, scal, x, c1w1, c1b1, N * 64);
  gemm(agg, c1w2, c1b2, h, N, 64, 64, /*relu*/1);
  k_bn<<<CEILDIV(N * 64, THREADS), THREADS, 0, stream>>>(h, c1bn, mask, /*bn then mask*/1, N, 64);

  // ---- GIN layers ----
  for (int i = 0; i < L; ++i) {
    k_scale_init<<<CEILDIV(N * 64, THREADS), THREADS, 0, stream>>>(agg, h, geps, i, N * 64);
    k_prop_feat<<<CEILDIV(E * 64, THREADS), THREADS, 0, stream>>>(row, col, h, agg, E);
    gemm(agg,  gw1 + (size_t)i * 64 * 64, gb1 + (size_t)i * 64, tbuf, N, 64, 64, 1);
    gemm(tbuf, gw2 + (size_t)i * 64 * 64, gb2 + (size_t)i * 64, agg,  N, 64, 64, 1);
    k_bn<<<CEILDIV(N * 64, THREADS), THREADS, 0, stream>>>(agg, gbn + (size_t)i * 4 * 64, nullptr, 0, N, 64);
    k_add<<<CEILDIV(N * 64, THREADS), THREADS, 0, stream>>>(h, agg, N * 64);
    prop_mask();
    k_bn<<<CEILDIV(N * 64, THREADS), THREADS, 0, stream>>>(h, bnsbn + (size_t)i * 4 * 64, mask, /*mask then bn*/2, N, 64);
  }

  // ---- GAT ----
  gemm(h, gatw, nullptr, xt, N, 64, 256, 0);
  k_gat_dots<<<CEILDIV(N * 8, THREADS), THREADS, 0, stream>>>(xt, gatt, asrc, adst, N);
  hipMemsetAsync(amax, 0, (size_t)N * 8 * sizeof(unsigned), stream);
  k_gat_alpha1<<<CEILDIV(Etot * 8, THREADS), THREADS, 0, stream>>>(row, col, asrc, adst, alpha, amax, E, Etot);
  hipMemsetAsync(denom, 0, (size_t)N * 8 * sizeof(float), stream);
  k_gat_alpha2<<<CEILDIV(Etot * 8, THREADS), THREADS, 0, stream>>>(row, alpha, amax, denom, E, Etot);
  k_gat_norm<<<CEILDIV(Etot * 8, THREADS), THREADS, 0, stream>>>(row, alpha, denom, E, Etot);
  hipMemsetAsync(gout, 0, (size_t)N * 256 * sizeof(float), stream);
  k_gat_agg<<<CEILDIV(Etot * 256, THREADS), THREADS, 0, stream>>>(row, col, alpha, xt, gout, E, Etot);

  prop_mask();
  k_bias_mask<<<CEILDIV(N * 256, THREADS), THREADS, 0, stream>>>(gout, gatb, mask, N);

  // ---- MLP head ----
  gemm(gout, l1w, l1b, tbuf, N, 256, 64, /*leaky 0.01*/2);
  k_bn<<<CEILDIV(N * 64, THREADS), THREADS, 0, stream>>>(tbuf, bn2, mask, /*mask then bn*/2, N, 64);
  k_lin2<<<CEILDIV(N, THREADS), THREADS, 0, stream>>>(tbuf, l2w, l2b, mask, hl, N);

  hipMemsetAsync(gmax, 0,    32 * sizeof(unsigned), stream);
  hipMemsetAsync(gmin, 0xFF, 32 * sizeof(unsigned), stream);
  k_group_minmax<<<CEILDIV(N, THREADS), THREADS, 0, stream>>>(hl, batch, gmax, gmin, N);
  k_final<<<CEILDIV(N, THREADS), THREADS, 0, stream>>>(hl, batch, gmax, gmin, mask, (float*)d_out, N);
}
